// DynamicExpansionBlock_26637387170279
// MI455X (gfx1250) — matrix operand order repaired and checked by reading, not compiled, not run on hardware
//
#include <hip/hip_runtime.h>
#include <hip/hip_bf16.h>
#include <cstdint>

typedef __attribute__((ext_vector_type(16))) _Float16 v16h;
typedef __attribute__((ext_vector_type(8)))  float    v8f;

#define TB  256
#define BM  128
#define BN  128
#define BK  32
#define LDK 40      // padded halves per LDS row (80B, 16B aligned, conflict-spread)

#define NB  32
#define SEQ 128
#define DM  1024
#define NE  16
#define NQ  2048    // SEQ*NE
#define KC  272     // 128 (P) + 128 (T) + 16 (S), multiple of 16

union FragU { v16h v; uint4 u[2]; };

__device__ inline v8f vzero8() { v8f z = {0.f,0.f,0.f,0.f,0.f,0.f,0.f,0.f}; return z; }

// Stage a BMxBK (rows x K) f16 tile into LDS. Rows clamped, K zero-filled.
__device__ inline void load_tile(const _Float16* __restrict__ src, size_t ld,
                                 int row0, int rows, int k0, int K,
                                 _Float16* __restrict__ lds) {
  const int t = threadIdx.x;
#pragma unroll
  for (int p = 0; p < 2; ++p) {
    int r  = p * 64 + (t >> 2);
    int c  = (t & 3) * 8;
    int gr = row0 + r; if (gr > rows - 1) gr = rows - 1;
    int gk = k0 + c;
    uint4 val{0u, 0u, 0u, 0u};
    if (gk < K) val = *(const uint4*)(src + (size_t)gr * ld + gk);
    *(uint4*)(&lds[r * LDK + c]) = val;
  }
}

// C[BMxBN] += A[M x K] * Bt[N x K]^T  (both row-major f16, f32 accumulate)
// 8 waves: wave (wm,wn) owns 32x64 = 2x4 tiles of 16x16.
__device__ inline void gemm_core(const _Float16* __restrict__ A, size_t lda, int M, int m0,
                                 const _Float16* __restrict__ Bt, size_t ldb, int N, int n0,
                                 int K, _Float16* lA, _Float16* lB, v8f acc[2][4]) {
  const int lane = threadIdx.x & 31;
  const int wave = threadIdx.x >> 5;
  const int wm = wave >> 1, wn = wave & 1;
  for (int kt = 0; kt < K; kt += BK) {
    __syncthreads();
    load_tile(A,  lda, m0, M, kt, K, lA);
    load_tile(Bt, ldb, n0, N, kt, K, lB);
    __syncthreads();
    FragU af[2], bf[4];
#pragma unroll
    for (int ti = 0; ti < 2; ++ti) {
      // A 16x32 f16 layout: lanes 0-15 hold K 0..7 / 16..23, lanes 16-31 K 8..15 / 24..31
      int r  = wm * 32 + ti * 16 + (lane & 15);
      int kb = (lane >> 4) * 8;
      af[ti].u[0] = *(const uint4*)(&lA[r * LDK + kb]);
      af[ti].u[1] = *(const uint4*)(&lA[r * LDK + kb + 16]);
    }
#pragma unroll
    for (int tj = 0; tj < 4; ++tj) {
      // B 32x16 f16 layout: lanes 0-15 hold K 0..7 / 16..23, lanes 16-31 K 8..15 / 24..31 (N = lane%16)
      int n  = wn * 64 + tj * 16 + (lane & 15);
      int kb = (lane >> 4) * 8;
      bf[tj].u[0] = *(const uint4*)(&lB[n * LDK + kb]);
      bf[tj].u[1] = *(const uint4*)(&lB[n * LDK + kb + 16]);
    }
#pragma unroll
    for (int ti = 0; ti < 2; ++ti)
#pragma unroll
      for (int tj = 0; tj < 4; ++tj)
        acc[ti][tj] = __builtin_amdgcn_wmma_f32_16x16x32_f16(
            false, af[ti].v, false, bf[tj].v, (short)0, acc[ti][tj], false, false);
  }
}

// ---------------- prep kernels ----------------

__global__ __launch_bounds__(TB) void k_cvt_x(const float* __restrict__ x,
                                              _Float16* __restrict__ xh) {
  size_t i = (size_t)blockIdx.x * TB + threadIdx.x;
  xh[i] = (_Float16)x[i];
}

__global__ __launch_bounds__(TB) void k_wt(const float* w0, const float* w1, const float* w2,
                                           const float* w3, const float* w4,
                                           _Float16* __restrict__ wt) {
  size_t i = (size_t)blockIdx.x * TB + threadIdx.x;   // 5 * 2^20
  int w = (int)(i >> 20);
  size_t r = i & 1048575u;
  int k = (int)(r >> 10), n = (int)(r & 1023);
  const float* W = (w == 0) ? w0 : (w == 1) ? w1 : (w == 2) ? w2 : (w == 3) ? w3 : w4;
  wt[((size_t)w << 20) + (size_t)n * DM + k] = (_Float16)W[(size_t)k * DM + n];
}

// gather query_emb rows into qc[b][0:16], bias_emb columns into BtA/BtB[...,256:272]
__global__ __launch_bounds__(TB) void k_gather(const float* __restrict__ qe,
                                               const float* __restrict__ be,
                                               const int* __restrict__ idx,
                                               _Float16* __restrict__ qc,
                                               _Float16* __restrict__ BtA,
                                               _Float16* __restrict__ BtB) {
  int i = blockIdx.x * TB + threadIdx.x;   // b*NE*DM
  int n = i & 1023;
  int r = i >> 10;                         // b*16+e
  int b = r >> 4, e = r & 15;
  int id = idx[r] & 15;
  qc[((size_t)b * 144 + e) * DM + n] = (_Float16)qe[(size_t)id * DM + n];
  _Float16 bv = (_Float16)be[(size_t)id * DM + n];
  size_t o = ((size_t)b * DM + n) * KC + 256 + e;
  BtA[o] = bv; BtB[o] = bv;
}

// ---------------- GEMM kernels ----------------

// Y = x @ W_w + b_w for w = 0..4; epilogue routes to the layout each consumer needs.
__global__ __launch_bounds__(TB) void k_gemm_xw(
    const _Float16* __restrict__ xh, const _Float16* __restrict__ wt,
    const float* b_cond, const float* b_key, const float* b_a,
    const float* b_b, const float* b_sel,
    _Float16* __restrict__ qc, _Float16* __restrict__ xkH, float* __restrict__ sel,
    _Float16* __restrict__ BtA, _Float16* __restrict__ BtB) {
  __shared__ _Float16 lA[BM * LDK], lB[BN * LDK];
  const int w  = blockIdx.z;
  const int m0 = blockIdx.y * BM;
  const int n0 = blockIdx.x * BN;
  v8f acc[2][4];
#pragma unroll
  for (int i = 0; i < 2; ++i)
#pragma unroll
    for (int j = 0; j < 4; ++j) acc[i][j] = vzero8();
  const _Float16* Bt = wt + ((size_t)w << 20);
  gemm_core(xh, DM, NB * SEQ, m0, Bt, DM, DM, n0, DM, lA, lB, acc);

  const float* bias = (w == 0) ? b_cond : (w == 1) ? b_key : (w == 2) ? b_a
                       : (w == 3) ? b_b : b_sel;
  const int lane = threadIdx.x & 31;
  const int wave = threadIdx.x >> 5;
  const int wm = wave >> 1, wn = wave & 1;
#pragma unroll
  for (int ti = 0; ti < 2; ++ti)
#pragma unroll
    for (int tj = 0; tj < 4; ++tj)
#pragma unroll
      for (int j = 0; j < 8; ++j) {
        int gm = m0 + wm * 32 + ti * 16 + (lane >> 4) * 8 + j;
        int gn = n0 + wn * 64 + tj * 16 + (lane & 15);
        float y = acc[ti][tj][j] + bias[gn];
        int b = gm >> 7, l = gm & 127;
        if (w == 0) {            // cond: qc rows 16..143 + transposed into BtA/BtB cols 128..255
          _Float16 h = (_Float16)y;
          qc[((size_t)b * 144 + 16 + l) * DM + gn] = h;
          size_t o = ((size_t)b * DM + gn) * KC + 128 + l;
          BtA[o] = h; BtB[o] = h;
        } else if (w == 1) {     // x_key: row-major (B side of QK^T)
          xkH[((size_t)b * SEQ + l) * DM + gn] = (_Float16)y;
        } else if (w == 2) {     // xa transposed into BtA cols 0..127
          BtA[((size_t)b * DM + gn) * KC + l] = (_Float16)y;
        } else if (w == 3) {     // xb transposed into BtB cols 0..127
          BtB[((size_t)b * DM + gn) * KC + l] = (_Float16)y;
        } else {                 // selector = sigmoid
          sel[(size_t)gm * DM + gn] = 1.f / (1.f + __expf(-y));
        }
      }
}

// GH[b][m][k] : rows 0..15 = G[e,k] = qv_e . xk_k ; rows 16..143 = H[l,k] = cond_l . xk_k
__global__ __launch_bounds__(TB) void k_gemm_gh(const _Float16* __restrict__ qc,
                                                const _Float16* __restrict__ xkH,
                                                float* __restrict__ GH) {
  __shared__ _Float16 lA[BM * LDK], lB[BN * LDK];
  const int b  = blockIdx.z;
  const int m0 = blockIdx.y * BM;
  v8f acc[2][4];
#pragma unroll
  for (int i = 0; i < 2; ++i)
#pragma unroll
    for (int j = 0; j < 4; ++j) acc[i][j] = vzero8();
  gemm_core(qc + (size_t)b * 144 * DM, DM, 144, m0,
            xkH + (size_t)b * SEQ * DM, DM, SEQ, 0, DM, lA, lB, acc);
  const int lane = threadIdx.x & 31;
  const int wave = threadIdx.x >> 5;
  const int wm = wave >> 1, wn = wave & 1;
#pragma unroll
  for (int ti = 0; ti < 2; ++ti)
#pragma unroll
    for (int tj = 0; tj < 4; ++tj)
#pragma unroll
      for (int j = 0; j < 8; ++j) {
        int gm = m0 + wm * 32 + ti * 16 + (lane >> 4) * 8 + j;
        int gn = wn * 64 + tj * 16 + (lane & 15);
        if (gm < 144 && gn < SEQ)
          GH[((size_t)b * 144 + gm) * SEQ + gn] = acc[ti][tj][j] * 0.03125f; // 1/sqrt(1024)
      }
}

// fw normalization: one block per (b, q) row of 128 keys; writes a_fw^T / c_fw^T (f16)
__global__ __launch_bounds__(128) void k_fw(const float* __restrict__ GH,
                                            const int* __restrict__ fwm,
                                            _Float16* __restrict__ afwT,
                                            _Float16* __restrict__ cfwT) {
  const int q = blockIdx.x, b = blockIdx.y, k = threadIdx.x;
  const int e = q & 15, l = q >> 4;
  float z = GH[((size_t)b * 144 + e) * SEQ + k] + GH[((size_t)b * 144 + 16 + l) * SEQ + k];
  float m = (fwm[((size_t)b * NQ + q) * SEQ + k] != 0) ? 1.f : 0.f;
  float a = fmaxf(z, 0.f) * m;
  float c = fmaxf(-z, 0.f) * m;
  __shared__ float ra[128], rc[128];
  ra[k] = a; rc[k] = c; __syncthreads();
  for (int s = 64; s > 0; s >>= 1) {
    if (k < s) { ra[k] += ra[k + s]; rc[k] += rc[k + s]; }
    __syncthreads();
  }
  size_t o = ((size_t)(b * SEQ + k)) * NQ + q;
  afwT[o] = (_Float16)(a / (ra[0] + 1e-9f));
  cfwT[o] = (_Float16)(c / (rc[0] + 1e-9f));
}

// bw normalization + S/T marginals: one block per (b, k) row of 2048 queries.
__global__ __launch_bounds__(TB) void k_bw(const float* __restrict__ GH,
                                           const int* __restrict__ bwm,
                                           _Float16* __restrict__ abw,
                                           _Float16* __restrict__ cbw,
                                           _Float16* __restrict__ Aa,
                                           _Float16* __restrict__ Ab) {
  const int k = blockIdx.x, b = blockIdx.y, t = threadIdx.x;
  __shared__ float reda[256], redc[256], Ta[128], Tc[128], Sa[16], Sc[16];
  if (t < 128) { Ta[t] = 0.f; Tc[t] = 0.f; }
  if (t < 16)  { Sa[t] = 0.f; Sc[t] = 0.f; }
  float av[8], cv[8], psa = 0.f, psc = 0.f;
#pragma unroll
  for (int j = 0; j < 8; ++j) {
    int q = t * 8 + j; int e = q & 15, l = q >> 4;
    float z = GH[((size_t)b * 144 + e) * SEQ + k] + GH[((size_t)b * 144 + 16 + l) * SEQ + k];
    float m = (bwm[((size_t)b * SEQ + k) * NQ + q] != 0) ? 1.f : 0.f;
    av[j] = fmaxf(z, 0.f) * m; cv[j] = fmaxf(-z, 0.f) * m;
    psa += av[j]; psc += cv[j];
  }
  reda[t] = psa; redc[t] = psc; __syncthreads();
  for (int s = 128; s > 0; s >>= 1) {
    if (t < s) { reda[t] += reda[t + s]; redc[t] += redc[t + s]; }
    __syncthreads();
  }
  const float inva = 1.f / (reda[0] + 1e-9f), invc = 1.f / (redc[0] + 1e-9f);
  float la = 0.f, lc = 0.f;
  const size_t rowo = ((size_t)b * SEQ + k) * NQ;
#pragma unroll
  for (int j = 0; j < 8; ++j) {
    int q = t * 8 + j; int e = q & 15;
    float afv = av[j] * inva, cfv = cv[j] * invc;
    abw[rowo + q] = (_Float16)afv; cbw[rowo + q] = (_Float16)cfv;
    la += afv; lc += cfv;
    atomicAdd(&Sa[e], afv); atomicAdd(&Sc[e], cfv);
  }
  int l = (t * 8) >> 4;   // 8 consecutive q share one l
  atomicAdd(&Ta[l], la); atomicAdd(&Tc[l], lc);
  __syncthreads();
  const size_t base = ((size_t)b * SEQ + k) * KC;
  if (t < 128) { Aa[base + 128 + t] = (_Float16)Ta[t]; Ab[base + 128 + t] = (_Float16)Tc[t]; }
  if (t < 16)  { Aa[base + 256 + t] = (_Float16)Sa[t]; Ab[base + 256 + t] = (_Float16)Sc[t]; }
}

// P = a_bw @ a_fw (128x128, K=2048) -> packed into Aa/Ab cols 0..127
__global__ __launch_bounds__(TB) void k_gemm_p(const _Float16* __restrict__ abw,
                                               const _Float16* __restrict__ cbw,
                                               const _Float16* __restrict__ afwT,
                                               const _Float16* __restrict__ cfwT,
                                               _Float16* __restrict__ Aa,
                                               _Float16* __restrict__ Ab) {
  __shared__ _Float16 lA[BM * LDK], lB[BN * LDK];
  const int id = blockIdx.z, b = id >> 1, path = id & 1;
  const _Float16* A  = (path ? cbw  : abw)  + (size_t)b * SEQ * NQ;
  const _Float16* Bt = (path ? cfwT : afwT) + (size_t)b * SEQ * NQ;
  _Float16* out = path ? Ab : Aa;
  v8f acc[2][4];
#pragma unroll
  for (int i = 0; i < 2; ++i)
#pragma unroll
    for (int j = 0; j < 4; ++j) acc[i][j] = vzero8();
  gemm_core(A, NQ, SEQ, 0, Bt, NQ, SEQ, 0, NQ, lA, lB, acc);
  const int lane = threadIdx.x & 31;
  const int wave = threadIdx.x >> 5;
  const int wm = wave >> 1, wn = wave & 1;
#pragma unroll
  for (int ti = 0; ti < 2; ++ti)
#pragma unroll
    for (int tj = 0; tj < 4; ++tj)
#pragma unroll
      for (int j = 0; j < 8; ++j) {
        int gm = wm * 32 + ti * 16 + (lane >> 4) * 8 + j;
        int gn = wn * 64 + tj * 16 + (lane & 15);
        if (gn < SEQ)
          out[((size_t)b * SEQ + gm) * KC + gn] = (_Float16)acc[ti][tj][j];
      }
}

// out = sel * (Aa@BtA^T) + (1-sel) * (Ab@BtB^T), K = 272
__global__ __launch_bounds__(TB) void k_final(const _Float16* __restrict__ Aa,
                                              const _Float16* __restrict__ Ab,
                                              const _Float16* __restrict__ BtA,
                                              const _Float16* __restrict__ BtB,
                                              const float* __restrict__ sel,
                                              float* __restrict__ out) {
  __shared__ _Float16 lA[BM * LDK], lB[BN * LDK];
  const int b  = blockIdx.z;
  const int n0 = blockIdx.x * BN;
  v8f accA[2][4], accB[2][4];
#pragma unroll
  for (int i = 0; i < 2; ++i)
#pragma unroll
    for (int j = 0; j < 4; ++j) { accA[i][j] = vzero8(); accB[i][j] = vzero8(); }
  gemm_core(Aa + (size_t)b * SEQ * KC, KC, SEQ, 0,
            BtA + (size_t)b * DM * KC, KC, DM, n0, KC, lA, lB, accA);
  gemm_core(Ab + (size_t)b * SEQ * KC, KC, SEQ, 0,
            BtB + (size_t)b * DM * KC, KC, DM, n0, KC, lA, lB, accB);
  const int lane = threadIdx.x & 31;
  const int wave = threadIdx.x >> 5;
  const int wm = wave >> 1, wn = wave & 1;
#pragma unroll
  for (int ti = 0; ti < 2; ++ti)
#pragma unroll
    for (int tj = 0; tj < 4; ++tj)
#pragma unroll
      for (int j = 0; j < 8; ++j) {
        int gm = wm * 32 + ti * 16 + (lane >> 4) * 8 + j;
        int gn = n0 + wn * 64 + tj * 16 + (lane & 15);
        size_t o = ((size_t)b * SEQ + gm) * DM + gn;
        float s = sel[o];
        out[o] = s * accA[ti][tj][j] + (1.f - s) * accB[ti][tj][j];
      }
}

// ---------------- host ----------------

extern "C" void kernel_launch(void* const* d_in, const int* in_sizes, int n_in,
                              void* d_out, int out_size, void* d_ws, size_t ws_size,
                              hipStream_t stream) {
  const float* x       = (const float*)d_in[0];
  const int*   n_idx   = (const int*)d_in[1];
  const int*   fw_mask = (const int*)d_in[2];
  const int*   bw_mask = (const int*)d_in[3];
  const float* W_cond  = (const float*)d_in[4];
  const float* b_cond  = (const float*)d_in[5];
  const float* q_emb   = (const float*)d_in[6];
  const float* bi_emb  = (const float*)d_in[7];
  const float* W_key   = (const float*)d_in[8];
  const float* b_key   = (const float*)d_in[9];
  const float* W_a     = (const float*)d_in[10];
  const float* b_a     = (const float*)d_in[11];
  const float* W_b     = (const float*)d_in[12];
  const float* b_b     = (const float*)d_in[13];
  const float* W_sel   = (const float*)d_in[14];
  const float* b_sel   = (const float*)d_in[15];
  float* out = (float*)d_out;

  char* ws = (char*)d_ws;
  size_t off = 0;
  auto take = [&](size_t bytes) { char* p = ws + off; off += (bytes + 255) & ~(size_t)255; return p; };

  _Float16* xh   = (_Float16*)take((size_t)NB * SEQ * DM * 2);   // x f16
  _Float16* wt   = (_Float16*)take((size_t)5 * DM * DM * 2);     // 5x W^T f16
  _Float16* qc   = (_Float16*)take((size_t)NB * 144 * DM * 2);   // [qv(16); cond(128)]
  _Float16* xkH  = (_Float16*)take((size_t)NB * SEQ * DM * 2);   // x_key rows f16
  float*    sel  = (float*)   take((size_t)NB * SEQ * DM * 4);   // sigmoid selector
  float*    GH   = (float*)   take((size_t)NB * 144 * SEQ * 4);  // G(16)+H(128) rows
  _Float16* afwT = (_Float16*)take((size_t)NB * SEQ * NQ * 2);   // a_fw^T [b][k][q]
  _Float16* cfwT = (_Float16*)take((size_t)NB * SEQ * NQ * 2);
  _Float16* abw  = (_Float16*)take((size_t)NB * SEQ * NQ * 2);   // a_bw rows [b][k][q]
  _Float16* cbw  = (_Float16*)take((size_t)NB * SEQ * NQ * 2);
  _Float16* Aa   = (_Float16*)take((size_t)NB * SEQ * KC * 2);   // [P|T|S] a-path
  _Float16* Ab   = (_Float16*)take((size_t)NB * SEQ * KC * 2);
  _Float16* BtA  = (_Float16*)take((size_t)NB * DM * KC * 2);    // [xa^T|cond^T|bvec^T]
  _Float16* BtB  = (_Float16*)take((size_t)NB * DM * KC * 2);
  (void)ws_size; (void)in_sizes; (void)n_in; (void)out_size;

  k_cvt_x<<<(NB * SEQ * DM) / TB, TB, 0, stream>>>(x, xh);
  k_wt<<<(5 * DM * DM) / TB, TB, 0, stream>>>(W_cond, W_key, W_a, W_b, W_sel, wt);
  k_gather<<<(NB * NE * DM) / TB, TB, 0, stream>>>(q_emb, bi_emb, n_idx, qc, BtA, BtB);

  // 5 projections: M=4096, N=1024, K=1024
  k_gemm_xw<<<dim3(DM / BN, (NB * SEQ) / BM, 5), TB, 0, stream>>>(
      xh, wt, b_cond, b_key, b_a, b_b, b_sel, qc, xkH, sel, BtA, BtB);

  // G/H: per batch M=144, N=128, K=1024
  k_gemm_gh<<<dim3(1, 2, NB), TB, 0, stream>>>(qc, xkH, GH);

  k_fw<<<dim3(NQ, NB), 128, 0, stream>>>(GH, fw_mask, afwT, cfwT);
  k_bw<<<dim3(SEQ, NB), TB, 0, stream>>>(GH, bw_mask, abw, cbw, Aa, Ab);

  // P = a_bw @ a_fw: per (batch, path) M=N=128, K=2048
  k_gemm_p<<<dim3(1, 1, 2 * NB), TB, 0, stream>>>(abw, cbw, afwT, cfwT, Aa, Ab);

  // final: per batch M=128, N=1024, K=272 (both paths fused + selector blend)
  k_final<<<dim3(DM / BN, 1, NB), TB, 0, stream>>>(Aa, Ab, BtA, BtB, sel, out);
}
